// RGCN_33423435498008
// MI455X (gfx1250) — matrix operand order repaired and checked by reading, not compiled, not run on hardware
//
#include <hip/hip_runtime.h>
#include <hip/hip_bf16.h>

// ---------------- problem constants (match reference) ----------------
#define NF_  100000
#define NA_  400
#define NC_  30
#define NN_  (NF_ + NA_ + NC_)   // 100430
#define D_   256

// B panel staging in LDS: 64 columns x 256 K bf16, padded column stride.
// 264 elements = 528 B = 132 dwords -> lane stride mod 64 banks = 4:
// a ds_load_b128 across 16 lanes covers banks {4i..4i+3}, conflict-minimal.
#define BSTRIDE 264

typedef __attribute__((ext_vector_type(16))) __bf16 v16bf;
typedef __attribute__((ext_vector_type(8)))  float  v8f;

union FragBF {
  uint4 q[2];   // 32 bytes
  v16bf v;      // 16 bf16
};

__device__ __forceinline__ unsigned short f2bf(float f) {
  unsigned int u = __float_as_uint(f);
  u += 0x7fffu + ((u >> 16) & 1u);      // round-to-nearest-even
  return (unsigned short)(u >> 16);
}

// ---------------------------------------------------------------------
// bf16 WMMA GEMM: C[rows,256] = A[rows,256] @ B[256,256] (+bias)
// A is bf16 row-major. Bt is bf16 with B pre-transposed: Bt[n*256 + k].
// Block = 256 threads = 8 wave32; block tile 256x64; wave tile 32x64.
// B panel (32 KB) is staged once per block into LDS via the gfx1250
// async-to-LDS path (ASYNCcnt), so the WMMA loop's global traffic is the
// A stream only (double-buffered); B fragments come from ds_load_b128.
// ---------------------------------------------------------------------
__global__ __launch_bounds__(256) void gemm_bf16_wmma(
    const unsigned short* __restrict__ A,
    const unsigned short* __restrict__ Bt,
    float* __restrict__ C,
    const float* __restrict__ bias,   // nullable
    int rows)
{
  __shared__ unsigned short bpanel[64 * BSTRIDE];   // 33,792 B

  const int tid  = threadIdx.x;
  const int wave = tid >> 5;
  const int lane = tid & 31;
  const int half = lane >> 4;
  const int l16  = lane & 15;

  const int n0 = blockIdx.y * 64;
  const int m0 = blockIdx.x * 256 + wave * 32;   // wave covers rows m0..m0+31

  // ---- async-stage the 64x256 B panel: 2048 x 16B chunks, 8 per thread ----
  // consecutive threads -> consecutive 16B chunks of one column (coalesced).
  {
    const unsigned short* gB = Bt + (size_t)n0 * 256;
    #pragma unroll
    for (int ii = 0; ii < 8; ii++) {
      const int ch    = tid + ii * 256;
      const int col   = ch >> 5;        // 0..63
      const int off16 = ch & 31;        // 16B chunk within the column
      const unsigned short* g = gB + col * 256 + off16 * 8;
      const unsigned ldsoff = (unsigned)(size_t)(&bpanel[col * BSTRIDE + off16 * 8]);
      asm volatile("global_load_async_to_lds_b128 %0, %1, off"
                   :: "v"(ldsoff), "v"(g) : "memory");
    }
  }

  // A fragment rows (ISA 7.12.2): lanes 0-15 and 16-31 both read rows M=0..15,
  // with K split by lane half; chunks at kb+half*8 and kb+16+half*8.
  const int rclamp = (rows > 0) ? rows - 1 : 0;
  const int arow0 = min(m0 + l16,      rclamp);
  const int arow1 = min(m0 + 16 + l16, rclamp);
  const unsigned short* ap0 = A + (size_t)arow0 * 256 + half * 8;
  const unsigned short* ap1 = A + (size_t)arow1 * 256 + half * 8;

  // wait for this wave's async transfers, then make panel visible to all waves
  asm volatile("s_wait_asynccnt 0x0" ::: "memory");
  __syncthreads();

  if (m0 >= rows) return;   // uniform per wave (after barrier): EXEC stays all-1s

  // B fragments from LDS: lane reads two 8-bf16 chunks of its column,
  // at k = half*8 and 16 + half*8 (matching the A fragment).
  const unsigned short* bl = &bpanel[(size_t)l16 * BSTRIDE + half * 8];

  v8f acc[8] = {};              // [j] = M-tile0 x N-tile j, [4+j] = M-tile1 x N-tile j
  FragBF fa0[2], fa1[2];

  auto load_a = [&](int kb, int buf) {
    fa0[buf].q[0] = *(const uint4*)(ap0 + kb);
    fa0[buf].q[1] = *(const uint4*)(ap0 + kb + 16);
    fa1[buf].q[0] = *(const uint4*)(ap1 + kb);
    fa1[buf].q[1] = *(const uint4*)(ap1 + kb + 16);
  };

  load_a(0, 0);
  #pragma unroll
  for (int it = 0; it < 8; ++it) {          // K = 8 steps of 32
    const int cur = it & 1;
    const int kb = it * 32;
    if (it < 7) load_a(kb + 32, cur ^ 1);   // prefetch next A step (global)
    #pragma unroll
    for (int j = 0; j < 4; j++) {
      FragBF fb;
      const unsigned short* bp = bl + j * (16 * BSTRIDE);
      fb.q[0] = *(const uint4*)(bp + kb);     // ds_load_b128
      fb.q[1] = *(const uint4*)(bp + kb + 16); // ds_load_b128
      acc[j]     = __builtin_amdgcn_wmma_f32_16x16x32_bf16(
          false, fa0[cur].v, false, fb.v, (short)0, acc[j],     false, false);
      acc[4 + j] = __builtin_amdgcn_wmma_f32_16x16x32_bf16(
          false, fa1[cur].v, false, fb.v, (short)0, acc[4 + j], false, false);
    }
  }

  float bv[4] = {0.f, 0.f, 0.f, 0.f};
  if (bias) {
    #pragma unroll
    for (int j = 0; j < 4; j++) bv[j] = bias[n0 + j * 16 + l16];
  }
  // C/D layout: VGPR i -> row (i + 8*half), col l16 within each 16x16 tile.
  #pragma unroll
  for (int i = 0; i < 8; i++) {
    const int r0 = m0 + half * 8 + i;
    if (r0 < rows) {
      float* cr = C + (size_t)r0 * 256 + n0 + l16;
      #pragma unroll
      for (int j = 0; j < 4; j++) cr[j * 16] = acc[j][i] + bv[j];
    }
    const int r1 = m0 + 16 + half * 8 + i;
    if (r1 < rows) {
      float* cr = C + (size_t)r1 * 256 + n0 + l16;
      #pragma unroll
      for (int j = 0; j < 4; j++) cr[j * 16] = acc[4 + j][i] + bv[j];
    }
  }
}

// ------------- per-type encoder: relu(x @ W + b) -> bf16 rows -------------
template <int KIN>
__global__ __launch_bounds__(256) void encoder_relu_bf16(
    const float* __restrict__ xin, const float* __restrict__ W,
    const float* __restrict__ b, unsigned short* __restrict__ xbf, int row_off)
{
  __shared__ float xs[KIN];
  const int n = blockIdx.x;
  const int c = threadIdx.x;
  if (c < KIN) xs[c] = xin[(size_t)n * KIN + c];
  __syncthreads();
  float acc = b[c];
  #pragma unroll
  for (int k = 0; k < KIN; k++) acc += xs[k] * W[k * 256 + c];
  acc = fmaxf(acc, 0.0f);
  xbf[(size_t)(row_off + n) * 256 + c] = f2bf(acc);
}

// ------ W_r = sum_b comp[r,b]*basis[b]; store bf16 TRANSPOSED [n][k] ------
// mat 0..3 -> relations, mat 4 -> root.  wt[mat][o*256 + i] = W[i][o]
__global__ __launch_bounds__(256) void prep_weights(
    const float* __restrict__ basis, const float* __restrict__ comp,
    const float* __restrict__ root, unsigned short* __restrict__ wt)
{
  int idx = blockIdx.x * 256 + threadIdx.x;   // 5 * 65536 threads
  int mat = idx >> 16;
  int io  = idx & 65535;
  int i = io >> 8, o = io & 255;
  float v;
  if (mat < 4) {
    v = comp[mat * 3 + 0] * basis[io]
      + comp[mat * 3 + 1] * basis[65536 + io]
      + comp[mat * 3 + 2] * basis[131072 + io];
  } else {
    v = root[io];
  }
  wt[(size_t)mat * 65536 + o * 256 + i] = f2bf(v);
}

// --------------------------- degree helpers ---------------------------
__global__ void zero_f32(float* __restrict__ v, int n) {
  int i = blockIdx.x * blockDim.x + threadIdx.x;
  if (i < n) v[i] = 0.0f;
}

__global__ void deg_count(const int* __restrict__ dst, float* __restrict__ deg,
                          int E, int dst_off) {
  int e = blockIdx.x * blockDim.x + threadIdx.x;
  if (e < E)
    __hip_atomic_fetch_add(&deg[dst[e] + dst_off], 1.0f,
                           __ATOMIC_RELAXED, __HIP_MEMORY_SCOPE_AGENT);
}

__global__ void inv_clamp(float* __restrict__ v, int n) {
  int i = blockIdx.x * blockDim.x + threadIdx.x;
  if (i < n) v[i] = 1.0f / fmaxf(v[i], 1.0f);
}

// ------- mean-aggregate scatter: out[dst] += y[src] * inv_deg[dst] -------
// 64 threads per edge, 4 channels each (float4 loads, f32 HW atomics)
__global__ __launch_bounds__(256) void scatter_mean(
    const float* __restrict__ y, const int* __restrict__ src,
    const int* __restrict__ dst, const float* __restrict__ inv_deg,
    float* __restrict__ out, int E, int dst_off)
{
  long long idx = (long long)blockIdx.x * blockDim.x + threadIdx.x;
  int e = (int)(idx >> 6);
  if (e >= E) return;
  int c = ((int)idx & 63) << 2;
  int s = src[e];
  int d = dst[e] + dst_off;
  float w = inv_deg[d];
  float4 v = *(const float4*)(y + (size_t)s * 256 + c);
  float* o = out + (size_t)d * 256 + c;
  __hip_atomic_fetch_add(o + 0, v.x * w, __ATOMIC_RELAXED, __HIP_MEMORY_SCOPE_AGENT);
  __hip_atomic_fetch_add(o + 1, v.y * w, __ATOMIC_RELAXED, __HIP_MEMORY_SCOPE_AGENT);
  __hip_atomic_fetch_add(o + 2, v.z * w, __ATOMIC_RELAXED, __HIP_MEMORY_SCOPE_AGENT);
  __hip_atomic_fetch_add(o + 3, v.w * w, __ATOMIC_RELAXED, __HIP_MEMORY_SCOPE_AGENT);
}

// --------- relu in place (f32) + produce bf16 activations for next layer ---------
__global__ void relu_to_bf16(float* __restrict__ out, unsigned short* __restrict__ xbf,
                             int n) {
  int i = blockIdx.x * blockDim.x + threadIdx.x;
  if (i < n) {
    float v = fmaxf(out[i], 0.0f);
    out[i] = v;
    xbf[i] = f2bf(v);
  }
}

// --------- readout: (h[:NF] @ W_out + b_out), one wave32 per row ---------
__global__ __launch_bounds__(256) void readout(
    const float* __restrict__ h, const float* __restrict__ Wout,
    const float* __restrict__ bout, float* __restrict__ out, int nf)
{
  int wave = threadIdx.x >> 5;
  int lane = threadIdx.x & 31;
  int n = blockIdx.x * 8 + wave;
  if (n >= nf) return;
  float acc = 0.0f;
  #pragma unroll
  for (int j = 0; j < 8; j++) {
    int c = lane + j * 32;
    acc += h[(size_t)n * 256 + c] * Wout[c];
  }
  #pragma unroll
  for (int off = 16; off > 0; off >>= 1)
    acc += __shfl_down(acc, off, 32);
  if (lane == 0) out[n] = acc + bout[0];
}

// ---------------------------------------------------------------------
extern "C" void kernel_launch(void* const* d_in, const int* in_sizes, int n_in,
                              void* d_out, int out_size, void* d_ws, size_t ws_size,
                              hipStream_t stream) {
  const float* x_flight = (const float*)d_in[0];
  const float* x_airport = (const float*)d_in[1];
  const float* x_carrier = (const float*)d_in[2];
  const float* W_f = (const float*)d_in[3];  const float* b_f = (const float*)d_in[4];
  const float* W_a = (const float*)d_in[5];  const float* b_a = (const float*)d_in[6];
  const float* W_c = (const float*)d_in[7];  const float* b_c = (const float*)d_in[8];
  const float* basisL[2] = {(const float*)d_in[9],  (const float*)d_in[13]};
  const float* compL[2]  = {(const float*)d_in[10], (const float*)d_in[14]};
  const float* rootL[2]  = {(const float*)d_in[11], (const float*)d_in[15]};
  const float* biasL[2]  = {(const float*)d_in[12], (const float*)d_in[16]};
  const float* W_out = (const float*)d_in[17];
  const float* b_out = (const float*)d_in[18];
  const int* srcR[4] = {(const int*)d_in[19], (const int*)d_in[21],
                        (const int*)d_in[23], (const int*)d_in[25]};
  const int* dstR[4] = {(const int*)d_in[20], (const int*)d_in[22],
                        (const int*)d_in[24], (const int*)d_in[26]};
  const int E = in_sizes[19];

  // relation metadata: source-type row base/rows in the concatenated node tensor,
  // destination global-index offset
  const int src_base[4] = {0,   NF_, 0,          NF_ + NA_};
  const int src_rows[4] = {NF_, NA_, NF_,        NC_};
  const int dst_off[4]  = {NF_, 0,   NF_ + NA_,  0};

  // ---- workspace carve-up (256B aligned) ----
  char* p = (char*)d_ws;
  auto take = [&](size_t bytes) -> void* {
    void* r = (void*)p;
    p += (bytes + 255) & ~(size_t)255;
    return r;
  };
  unsigned short* x_bf   = (unsigned short*)take((size_t)NN_ * 256 * 2); // bf16 activations
  float*          outb   = (float*)take((size_t)NN_ * 256 * 4);          // f32 layer output
  float*          y      = (float*)take((size_t)NF_ * 256 * 4);          // relation messages
  unsigned short* wt     = (unsigned short*)take((size_t)5 * 65536 * 2); // bf16 W_r^T + root^T
  float*          invdeg = (float*)take((size_t)4 * NN_ * 4);            // per-relation 1/deg

  // ---- encoders (f32 compute, bf16 output, relu fused) ----
  encoder_relu_bf16<32><<<NF_, 256, 0, stream>>>(x_flight,  W_f, b_f, x_bf, 0);
  encoder_relu_bf16<16><<<NA_, 256, 0, stream>>>(x_airport, W_a, b_a, x_bf, NF_);
  encoder_relu_bf16<8 ><<<NC_, 256, 0, stream>>>(x_carrier, W_c, b_c, x_bf, NF_ + NA_);

  // ---- per-relation inverse mean degree (edge lists fixed across layers) ----
  {
    int n = 4 * NN_;
    zero_f32<<<(n + 255) / 256, 256, 0, stream>>>(invdeg, n);
    for (int r = 0; r < 4; r++)
      deg_count<<<(E + 255) / 256, 256, 0, stream>>>(dstR[r], invdeg + (size_t)r * NN_,
                                                     E, dst_off[r]);
    inv_clamp<<<(n + 255) / 256, 256, 0, stream>>>(invdeg, n);
  }

  const int scatter_blocks = (int)(((long long)E * 64 + 255) / 256);

  // ---- two RGCN layers ----
  for (int l = 0; l < 2; l++) {
    prep_weights<<<(5 * 65536) / 256, 256, 0, stream>>>(basisL[l], compL[l], rootL[l], wt);

    // root/self transform writes the full out buffer: out = x @ root + bias
    gemm_bf16_wmma<<<dim3((NN_ + 255) / 256, 4), 256, 0, stream>>>(
        x_bf, wt + (size_t)4 * 65536, outb, biasL[l], NN_);

    // per relation: y = x_src_type @ W_r, then out[dst] += y[src] / deg[dst]
    for (int r = 0; r < 4; r++) {
      gemm_bf16_wmma<<<dim3((src_rows[r] + 255) / 256, 4), 256, 0, stream>>>(
          x_bf + (size_t)src_base[r] * 256, wt + (size_t)r * 65536, y,
          nullptr, src_rows[r]);
      scatter_mean<<<scatter_blocks, 256, 0, stream>>>(
          y, srcR[r], dstR[r], invdeg + (size_t)r * NN_, outb, E, dst_off[r]);
    }

    // relu in place + bf16 activations for next layer
    int n = NN_ * 256;
    relu_to_bf16<<<(n + 255) / 256, 256, 0, stream>>>(outb, x_bf, n);
  }

  // ---- readout over flight nodes ----
  readout<<<(NF_ + 7) / 8, 256, 0, stream>>>(outb, W_out, b_out, (float*)d_out, NF_);
}
